// ConformalModelLogits_2980707304085
// MI455X (gfx1250) — compile-verified, matrix operand order checked by reading, not run on hardware
//
#include <hip/hip_runtime.h>

#define NTHR 256
#define NPAD 1024
#define EPT  4   // NPAD / NTHR

typedef float    v4f  __attribute__((ext_vector_type(4)));
typedef float    v8f  __attribute__((ext_vector_type(8)));
typedef _Float16 v16h __attribute__((ext_vector_type(16)));
typedef int      v4i  __attribute__((ext_vector_type(4)));

typedef __attribute__((address_space(1))) v4i gv4i;  // global 16B vector
typedef __attribute__((address_space(3))) v4i lv4i;  // LDS 16B vector

#if defined(__has_builtin)
  #if __has_builtin(__builtin_amdgcn_global_load_async_to_lds_b128)
    #define HAVE_ASYNC128 1
  #endif
  #if __has_builtin(__builtin_amdgcn_s_wait_asynccnt)
    #define HAVE_WAIT_ASYNC 1
  #endif
#endif
#ifndef HAVE_ASYNC128
#define HAVE_ASYNC128 0
#endif
#ifndef HAVE_WAIT_ASYNC
#define HAVE_WAIT_ASYNC 0
#endif

// Monotone (order-preserving) float -> u32 map; we sort DESCENDING on this.
__device__ __forceinline__ unsigned fkeyu(float f) {
  unsigned b = __float_as_uint(f);
  return (b & 0x80000000u) ? ~b : (b | 0x80000000u);
}
__device__ __forceinline__ float ukeyf(unsigned u) {
  unsigned b = (u & 0x80000000u) ? (u & 0x7fffffffu) : ~u;
  return __uint_as_float(b);
}

// One-time inclusive scan of the penalty vector into workspace.
__global__ void pcs_scan_kernel(const float* __restrict__ pen,
                                float* __restrict__ pcs, int C) {
  if (threadIdx.x == 0 && blockIdx.x == 0) {
    float run = 0.0f;
    for (int c = 0; c < C; ++c) { run += pen[c]; pcs[c] = run; }
  }
}

__global__ __launch_bounds__(NTHR) void conformal_kernel(
    const float* __restrict__ logits, const float* __restrict__ pcs,
    const float* __restrict__ uvec, const float* __restrict__ Tptr,
    const float* __restrict__ Qptr, float* __restrict__ outLogits,
    float* __restrict__ outMask, float* __restrict__ outSizes, int C) {
  __shared__ float              lvals[NPAD];   // row logits, original order (-inf pad)
  __shared__ unsigned long long keys[NPAD];    // sort keys (value-bits | ~index)
  __shared__ float              ordp[NPAD];    // sorted probabilities
  __shared__ float              csum[NPAD];    // inclusive cumsum of sorted probs
  __shared__ _Float16           parth[2 * NTHR];  // f16 partial counts (upper half 0)
  __shared__ float              wred[8];       // per-wave reduction slots
  __shared__ float              sCountF;
  __shared__ unsigned long long sBkey;

  const int    tid     = threadIdx.x;
  const int    lane    = tid & 31;
  const int    wid     = tid >> 5;
  const int    row     = blockIdx.x;
  const size_t rowBase = (size_t)row * (size_t)C;
  const float  invT    = 1.0f / Tptr[0];
  const float  tau     = Qptr[0];

  // ---------------- stage row into LDS (async global->LDS copy) --------------
  const float* grow = logits + rowBase;
#if HAVE_ASYNC128
  {
    const int nv = C >> 2;
    for (int i = tid; i < nv; i += NTHR) {
      __builtin_amdgcn_global_load_async_to_lds_b128(
          (gv4i*)(grow + 4 * i), (lv4i*)&lvals[4 * i], 0, 0);
    }
    for (int c = (nv << 2) + tid; c < C; c += NTHR) lvals[c] = grow[c];
#if HAVE_WAIT_ASYNC
    __builtin_amdgcn_s_wait_asynccnt(0);
#else
    asm volatile("s_wait_asynccnt 0" ::: "memory");
#endif
  }
#else
  for (int c = tid; c < C; c += NTHR) lvals[c] = grow[c];
#endif
  for (int c = C + tid; c < NPAD; c += NTHR) lvals[c] = -__builtin_inff();
  parth[NTHR + tid] = (_Float16)0.0f;  // zero the pad half once, up front
  __syncthreads();

  const int base = tid * EPT;

  // ---------------- row max (stable softmax) ---------------------------------
  float mx = -__builtin_inff();
#pragma unroll
  for (int j = 0; j < EPT; ++j) mx = fmaxf(mx, lvals[base + j]);
#pragma unroll
  for (int d = 16; d > 0; d >>= 1) mx = fmaxf(mx, __shfl_xor(mx, d, 32));
  if (lane == 0) wred[wid] = mx;
  __syncthreads();
  mx = wred[0];
#pragma unroll
  for (int w = 1; w < 8; ++w) mx = fmaxf(mx, wred[w]);
  __syncthreads();

  // ---------------- softmax denominator Z (precise f32 reduce) ---------------
  float zp = 0.0f;
#pragma unroll
  for (int j = 0; j < EPT; ++j) zp += expf((lvals[base + j] - mx) * invT);
#pragma unroll
  for (int d = 16; d > 0; d >>= 1) zp += __shfl_xor(zp, d, 32);
  if (lane == 0) wred[wid] = zp;
  __syncthreads();
  float Z = 0.0f;
#pragma unroll
  for (int w = 0; w < 8; ++w) Z += wred[w];
  const float invZ = 1.0f / Z;
  __syncthreads();  // wred reused below

  // ---------------- build unique 64-bit sort keys ----------------------------
#pragma unroll
  for (int j = 0; j < EPT; ++j) {
    int k = base + j;
    keys[k] = ((unsigned long long)fkeyu(lvals[k]) << 32) |
              (unsigned)(0xFFFFFFFFu - (unsigned)k);  // tie-break: small idx first
  }

  // ---------------- bitonic sort, descending, 1024 x u64 in LDS --------------
  for (unsigned kk = 2; kk <= NPAD; kk <<= 1) {
    for (unsigned jj = kk >> 1; jj > 0; jj >>= 1) {
      __syncthreads();
#pragma unroll
      for (int e = 0; e < EPT; ++e) {
        unsigned i   = (unsigned)tid + (unsigned)(e * NTHR);
        unsigned ixj = i ^ jj;
        if (ixj > i) {
          unsigned long long a = keys[i];
          unsigned long long b = keys[ixj];
          bool up = ((i & kk) == 0);
          if ((a < b) == up) { keys[i] = b; keys[ixj] = a; }
        }
      }
    }
  }
  __syncthreads();

  // ---------------- sorted probs + inclusive cumsum (two-level scan) ---------
  float pv[EPT], incl[EPT];
  float run = 0.0f;
#pragma unroll
  for (int j = 0; j < EPT; ++j) {
    unsigned hi = (unsigned)(keys[base + j] >> 32);
    float p = expf((ukeyf(hi) - mx) * invT) * invZ;  // pads: exp(-inf)=0
    run += p; pv[j] = p; incl[j] = run;
  }
  float x = run;
#pragma unroll
  for (int d = 1; d < 32; d <<= 1) {
    float y = __shfl_up(x, d, 32);
    if (lane >= d) x += y;
  }
  if (lane == 31) wred[wid] = x;
  __syncthreads();
  float woff = 0.0f;
  for (int w = 0; w < wid; ++w) woff += wred[w];
  const float excl = woff + (x - run);
#pragma unroll
  for (int j = 0; j < EPT; ++j) {
    csum[base + j] = excl + incl[j];
    ordp[base + j] = pv[j];
  }
  __syncthreads();

  // ---------------- threshold count: exact f16 WMMA reduction ----------------
  float cf = 0.0f;
#pragma unroll
  for (int j = 0; j < EPT; ++j) {
    int k = base + j;
    if (k < C && (csum[k] + pcs[k] <= tau)) cf += 1.0f;
  }
  parth[tid] = (_Float16)cf;  // integers in [0,4]: exact in f16; sum < 2048: exact
  __syncthreads();
  if (tid < 32) {  // full wave 0 -> EXEC all ones, as WMMA requires
    // Lane l carries 16 consecutive f16 partials: 32 lanes x 16 = 512 slots
    // (upper 256 are the zero pad), so every partial enters A exactly once.
    const v16h av = *(const v16h*)&parth[lane * 16];  // two ds_load_b128
    v16h ov;
#pragma unroll
    for (int q = 0; q < 16; ++q) ov[q] = (_Float16)1.0f;
    v8f acc = {0.0f, 0.0f, 0.0f, 0.0f, 0.0f, 0.0f, 0.0f, 0.0f};
    // D = A x ones + 0: every column of D equals the vector of A-row sums,
    // so sum(D) = 16 * sum(A) independent of the C/D VGPR striping.
    acc = __builtin_amdgcn_wmma_f32_16x16x32_f16(false, av, false, ov,
                                                 (short)0, acc, false, false);
    float s = acc[0] + acc[1] + acc[2] + acc[3] +
              acc[4] + acc[5] + acc[6] + acc[7];
#pragma unroll
    for (int d = 16; d > 0; d >>= 1) s += __shfl_xor(s, d, 32);
    if (lane == 0) sCountF = s * 0.0625f;  // /16: exact (integer * 2^-4)
  }
  __syncthreads();

  // ---------------- randomized set size ---------------------------------------
  if (tid == 0) {
    int m  = (int)(sCountF + 0.5f);
    int sb = m + 1;
    if (sb > C) sb = C;
    float ord_s = ordp[sb - 1];
    float cum_s = csum[sb - 1];
    float pcs_s = pcs[sb - 1];
    float V  = (tau - (cum_s - ord_s) - pcs_s) / ord_s;
    int   sz = sb - ((uvec[row] >= V) ? 1 : 0);
    if (tau == 1.0f) sz = C;
    if (sz < 1) sz = 1;
    sBkey = keys[sz - 1];
    outSizes[row] = (float)sz;
  }
  __syncthreads();

  // ---------------- coalesced outputs: logits passthrough + membership mask ---
  const unsigned long long bkey = sBkey;  // keys unique -> exactly `sz` pass >=
  const int nv4 = C >> 2;
  if (tid < nv4) {
    int c0 = tid << 2;
    v4f lv, mk;
#pragma unroll
    for (int j = 0; j < 4; ++j) {
      float f = lvals[c0 + j];
      lv[j]   = f;
      unsigned long long kk2 =
          ((unsigned long long)fkeyu(f) << 32) |
          (unsigned)(0xFFFFFFFFu - (unsigned)(c0 + j));
      mk[j] = (kk2 >= bkey) ? 1.0f : 0.0f;
    }
    __builtin_nontemporal_store(lv, (v4f*)(outLogits + rowBase + c0));
    __builtin_nontemporal_store(mk, (v4f*)(outMask + rowBase + c0));
  }
  for (int c = (nv4 << 2) + tid; c < C; c += NTHR) {  // tail (empty for C=1000)
    float f = lvals[c];
    unsigned long long kk2 = ((unsigned long long)fkeyu(f) << 32) |
                             (unsigned)(0xFFFFFFFFu - (unsigned)c);
    outLogits[rowBase + c] = f;
    outMask[rowBase + c]   = (kk2 >= bkey) ? 1.0f : 0.0f;
  }
}

extern "C" void kernel_launch(void* const* d_in, const int* in_sizes, int n_in,
                              void* d_out, int out_size, void* d_ws, size_t ws_size,
                              hipStream_t stream) {
  const float* logits = (const float*)d_in[0];
  const float* pen    = (const float*)d_in[1];
  const float* u      = (const float*)d_in[2];
  const float* T      = (const float*)d_in[3];
  const float* Qhat   = (const float*)d_in[4];
  const int Cn = in_sizes[1];  // penalties: [1, C]
  const int Bn = in_sizes[2];  // u: [B]

  float* out       = (float*)d_out;
  float* outLogits = out;
  float* outMask   = out + (size_t)Bn * (size_t)Cn;
  float* outSizes  = out + 2 * (size_t)Bn * (size_t)Cn;
  float* pcs       = (float*)d_ws;  // C floats of scratch

  pcs_scan_kernel<<<1, 64, 0, stream>>>(pen, pcs, Cn);
  conformal_kernel<<<dim3(Bn), dim3(NTHR), 0, stream>>>(
      logits, pcs, u, T, Qhat, outLogits, outMask, outSizes, Cn);
}
